// BiLSTM_12257836663215
// MI455X (gfx1250) — compile-verified, matrix operand order checked by reading, not run on hardware
//
#include <hip/hip_runtime.h>
#include <stdint.h>

// ---------------------------------------------------------------------------
// BiLSTM persistent-scan kernel for gfx1250 (MI455X).
//  - rows of the recurrence are independent (transpose quirk) -> partition
//    M=128 rows over 16 workgroups (8 per direction), no cross-WG sync.
//  - bf16 WMMA (v_wmma_f32_16x16x32_bf16), f32 accumulate.
//  - weights repacked once into fragment-major bf16 in d_ws (1.5 MB, L2-resident).
//  - K-tiles 0..2 of B kept VGPR-resident (96 VGPRs/wave); K-tiles 3..11
//    streamed from L2 each step, software-pipelined with double-buffered
//    4-fragment chunks (issue chunk s+2 after consuming chunk s) so L2
//    latency overlaps WMMA. Visit order rotated by (t mod 9) so the compiler
//    cannot hoist the streamed loads (which previously caused scratch spills).
//  - A matrix (16x384) double-buffered in LDS; h written in place by each wave;
//    one workgroup barrier per timestep.
// ---------------------------------------------------------------------------

#define S_LEN 1024
#define B_SZ  128
#define D_SZ  128
#define H_SZ  256
#define KDIM  384           // D + H
#define KT_N  12            // 384 / 32
#define KT_RES 3            // K-tiles kept resident in VGPRs
#define KT_STR 9            // K-tiles streamed per step (3..11)
#define NT_N  64            // 1024 / 16  (4 gates * 256 cols)
#define THREADS 512         // 16 waves (wave32)

typedef __attribute__((ext_vector_type(16))) __bf16       v16bf;
typedef __attribute__((ext_vector_type(8)))  float        v8f;
typedef __attribute__((ext_vector_type(4)))  unsigned int v4u;

union Frag { v16bf v; v4u u[2]; };

__device__ __forceinline__ unsigned short f32_bf16(float f) {
  unsigned int u = __builtin_bit_cast(unsigned int, f);
  u += 0x7FFFu + ((u >> 16) & 1u);          // round-to-nearest-even
  return (unsigned short)(u >> 16);
}

// 16x32 bf16 fragment layout (ISA 7.12.2): lane-group from kk, half index hh.
// Forward: kk = (vg<4?0:16) + lg*8 + (vg&3)*2 + within, vg = hh>>1.
__device__ __forceinline__ void kk_to_pos(int kk, int& lgsel, int& hh) {
  lgsel  = (kk >> 3) & 1;
  int vg = ((kk & 16) ? 4 : 0) + ((kk & 7) >> 1);
  hh     = vg * 2 + (kk & 1);
}

// ---------------------------------------------------------------------------
// Weight repack: (dir, gate g, j, k) f32 -> fragment-major bf16 blocks.
// Block (dir, nt, kt) = 1024 B laid out as [c(2)][lane(32)][8 halves],
// so the main kernel loads each half-fragment with one fully-coalesced b128.
// ---------------------------------------------------------------------------
__global__ void prep_w_kernel(const float* __restrict__ Wf,
                              const float* __restrict__ Wb,
                              unsigned short* __restrict__ ws) {
  int idx = blockIdx.x * blockDim.x + threadIdx.x;     // one bf16 element each
  const int total = 2 * NT_N * KT_N * 512;
  if (idx >= total) return;
  int within = idx & 511;                               // element inside block
  int blk    = idx >> 9;
  int kt  = blk % KT_N;
  int rem = blk / KT_N;
  int nt  = rem & 63;
  int dir = rem >> 6;
  int h8   = within & 7;
  int lane = (within >> 3) & 31;
  int c    = within >> 8;
  int hh   = c * 8 + h8;
  int lg   = lane >> 4;
  int nn   = lane & 15;
  int vg   = hh >> 1;
  int kk   = ((vg < 4) ? 0 : 16) + lg * 8 + (vg & 3) * 2 + (hh & 1);
  int n = nt * 16 + nn;                                 // 0..1023 = g*256 + j
  int g = n >> 8, j = n & 255;
  int k = kt * 32 + kk;
  const float* W = dir ? Wb : Wf;
  ws[idx] = f32_bf16(W[(g * H_SZ + j) * KDIM + k]);
}

// Cooperative build of the x-part of A (ktiles 0..3) for time index tx.
// A[i][k] = x[k, tx, r0+i]; 512 threads, each handles 4 k values of one column.
__device__ __forceinline__ void write_x_part(const float* __restrict__ x,
                                             int tx, int dirstep, int r0,
                                             unsigned char* abuf, int tid) {
  int i  = tid & 15;
  int kb = (tid >> 4) * 4;
  #pragma unroll
  for (int q = 0; q < 4; ++q) {
    int k = kb + q;
    const float* xp = &x[((size_t)k * S_LEN + tx) * D_SZ + r0 + i];
    float v = *xp;
    __builtin_prefetch(xp + dirstep, 0, 1);             // next step's x slice
    int lgsel, hh; kk_to_pos(k & 31, lgsel, hh);
    int lp = lgsel * 16 + i;
    *reinterpret_cast<unsigned short*>(abuf + (k >> 5) * 1024 + lp * 32 + hh * 2)
        = f32_bf16(v);
  }
}

__global__ __launch_bounds__(THREADS)
void bilstm_scan_kernel(const float* __restrict__ x,
                        const unsigned short* __restrict__ wsW,
                        const float* __restrict__ b_fwd,
                        const float* __restrict__ b_bwd,
                        float* __restrict__ out) {
  // Double-buffered A (16 rows x 384 K, bf16, fragment-major): 2 x 12 KB.
  __shared__ __align__(32) unsigned char Abuf[2 * KT_N * 1024];

  const int tid  = threadIdx.x;
  const int w    = tid >> 5;          // wave id 0..15 == jtile
  const int lane = tid & 31;
  const int bid  = blockIdx.x;
  const int dir  = bid >> 3;          // 0 = fwd, 1 = bwd
  const int r0   = (bid & 7) * 16;    // row block
  const int lg   = lane >> 4;
  const int jcol = w * 16 + (lane & 15);     // hidden column owned by lane
  const int dirstep = dir ? -D_SZ : D_SZ;
  const float* bias = dir ? b_bwd : b_fwd;

  // h-write position inside A buffer (k = 128 + jcol is fixed per lane)
  const int kh = 128 + jcol;
  int hlgsel, hhh; kk_to_pos(kh & 31, hlgsel, hhh);
  const int kth = kh >> 5;

  float c_state[8];
  #pragma unroll
  for (int r = 0; r < 8; ++r) c_state[r] = 0.f;
  float bgate[4];
  #pragma unroll
  for (int q = 0; q < 4; ++q) bgate[q] = bias[q * H_SZ + jcol];

  // Per-gate per-lane weight base pointers (fragment blocks are 1024 B).
  const unsigned char* gq[4];
  #pragma unroll
  for (int q = 0; q < 4; ++q)
    gq[q] = reinterpret_cast<const unsigned char*>(wsW) +
            (size_t)((dir * NT_N + q * 16 + w) * KT_N) * 1024 + lane * 16;

  // ---- resident B fragments: K-tiles 0..2, 96 VGPRs ----
  Frag bres[4][KT_RES];
  #pragma unroll
  for (int q = 0; q < 4; ++q)
    #pragma unroll
    for (int kt = 0; kt < KT_RES; ++kt) {
      bres[q][kt].u[0] = *reinterpret_cast<const v4u*>(gq[q] + (kt << 10));
      bres[q][kt].u[1] = *reinterpret_cast<const v4u*>(gq[q] + (kt << 10) + 512);
    }

  // ---- prologue: h(t=0)=0, x-part for first scan step ----
  const v4u zero4 = {0u, 0u, 0u, 0u};
  for (int off = tid * 16; off < 8192; off += THREADS * 16)
    *reinterpret_cast<v4u*>(&Abuf[4 * 1024 + off]) = zero4;   // ktiles 4..11 of buf0
  write_x_part(x, dir ? (S_LEN - 1) : 0, dirstep, r0, &Abuf[0], tid);
  __syncthreads();

  const size_t hBase = (size_t)B_SZ * S_LEN * 2 * H_SZ;       // 67108864
  const size_t cBase = hBase + (size_t)2 * B_SZ * H_SZ;

  for (int t = 0; t < S_LEN; ++t) {
    const int cur = t & 1, nxt = cur ^ 1;
    const unsigned char* Ab = &Abuf[cur * (KT_N * 1024)];
    const int rot = t % KT_STR;                 // streamed visit rotation

    v8f acc[4];
    #pragma unroll
    for (int q = 0; q < 4; ++q)
      acc[q] = (v8f){0.f, 0.f, 0.f, 0.f, 0.f, 0.f, 0.f, 0.f};

    // Streamed-chunk helpers (ktp(s) = 3 + (rot + s) mod 9).
    Frag bbuf[2][4];
    #define KTP(s)  (KT_RES + ((rot + (s) >= KT_STR) ? rot + (s) - KT_STR : rot + (s)))
    #define ISSUE(slot, s) do {                                               \
      const int _off = KTP(s) << 10;                                          \
      _Pragma("unroll")                                                       \
      for (int _q = 0; _q < 4; ++_q) {                                        \
        bbuf[slot][_q].u[0] = *reinterpret_cast<const v4u*>(gq[_q] + _off);   \
        bbuf[slot][_q].u[1] = *reinterpret_cast<const v4u*>(gq[_q] + _off + 512); \
      }                                                                       \
    } while (0)

    // Prime the pipeline: chunks 0 and 1 in flight before resident WMMAs.
    ISSUE(0, 0);
    ISSUE(1, 1);

    // ---- GEMM part 1: resident K-tiles 0..2 (hides first L2 round-trip) ----
    #pragma unroll
    for (int kt = 0; kt < KT_RES; ++kt) {
      Frag af;
      const unsigned char* ap = Ab + kt * 1024 + lane * 32;
      af.u[0] = *reinterpret_cast<const v4u*>(ap);
      af.u[1] = *reinterpret_cast<const v4u*>(ap + 16);
      #pragma unroll
      for (int q = 0; q < 4; ++q)
        acc[q] = __builtin_amdgcn_wmma_f32_16x16x32_bf16(
            false, af.v, false, bres[q][kt].v, (short)0, acc[q], false, false);
    }

    // ---- GEMM part 2: streamed K-tiles, double-buffered pipeline ----
    #pragma unroll
    for (int s = 0; s < KT_STR; ++s) {
      Frag af;
      const unsigned char* ap = Ab + (KTP(s) << 10) + lane * 32;
      af.u[0] = *reinterpret_cast<const v4u*>(ap);
      af.u[1] = *reinterpret_cast<const v4u*>(ap + 16);
      #pragma unroll
      for (int q = 0; q < 4; ++q)
        acc[q] = __builtin_amdgcn_wmma_f32_16x16x32_bf16(
            false, af.v, false, bbuf[s & 1][q].v, (short)0, acc[q], false, false);
      if (s + 2 < KT_STR) ISSUE(s & 1, s + 2);   // refill consumed slot
    }
    #undef ISSUE
    #undef KTP

    // ---- LSTM gate update (register-local: all 4 gates for this j-slice) ----
    float hval[8];
    #pragma unroll
    for (int r = 0; r < 8; ++r) {
      float zf = acc[0][r] + bgate[0];
      float zi = acc[1][r] + bgate[1];
      float zc = acc[2][r] + bgate[2];
      float zo = acc[3][r] + bgate[3];
      float fg = __frcp_rn(1.f + __expf(-zf));
      float ig = __frcp_rn(1.f + __expf(-zi));
      float og = __frcp_rn(1.f + __expf(-zo));
      float ct = 1.f - 2.f * __frcp_rn(__expf(2.f * zc) + 1.f);   // tanh
      float cn = fg * c_state[r] + ig * ct;
      c_state[r] = cn;
      hval[r] = og * (1.f - 2.f * __frcp_rn(__expf(2.f * cn) + 1.f));
    }

    // ---- write output h to global (B, S, 2H), time-reversed for bwd ----
    const int tout = dir ? (S_LEN - 1 - t) : t;
    #pragma unroll
    for (int r = 0; r < 8; ++r) {
      int m = r + 8 * lg;
      out[(size_t)(r0 + m) * S_LEN * 512 + (size_t)tout * 512 + dir * 256 + jcol]
          = hval[r];
    }
    if (t == S_LEN - 1) {                          // final states (2,B,H)
      #pragma unroll
      for (int r = 0; r < 8; ++r) {
        int m = r + 8 * lg;
        size_t si = (size_t)dir * B_SZ * H_SZ + (size_t)(r0 + m) * H_SZ + jcol;
        out[hBase + si] = hval[r];
        out[cBase + si] = c_state[r];
      }
    }

    // ---- build A for next step into the other buffer ----
    unsigned char* An = &Abuf[nxt * (KT_N * 1024)];
    #pragma unroll
    for (int r = 0; r < 8; ++r) {                  // h part (this wave's jcols)
      int m  = r + 8 * lg;
      int lp = hlgsel * 16 + m;
      *reinterpret_cast<unsigned short*>(An + kth * 1024 + lp * 32 + hhh * 2)
          = f32_bf16(hval[r]);
    }
    if (t + 1 < S_LEN) {
      int tx = dir ? (S_LEN - 2 - t) : (t + 1);
      write_x_part(x, tx, dirstep, r0, An, tid);
    }
    __syncthreads();                               // one barrier per timestep
  }
}

extern "C" void kernel_launch(void* const* d_in, const int* in_sizes, int n_in,
                              void* d_out, int out_size, void* d_ws, size_t ws_size,
                              hipStream_t stream) {
  (void)in_sizes; (void)n_in; (void)out_size; (void)ws_size;
  const float* x     = (const float*)d_in[0];
  const float* W_fwd = (const float*)d_in[1];
  const float* b_fwd = (const float*)d_in[2];
  const float* W_bwd = (const float*)d_in[3];
  const float* b_bwd = (const float*)d_in[4];
  float* out = (float*)d_out;
  unsigned short* wsW = (unsigned short*)d_ws;     // 1.5 MB bf16 fragment store

  const int totalW = 2 * NT_N * KT_N * 512;        // 786432 elements
  prep_w_kernel<<<totalW / 256, 256, 0, stream>>>(W_fwd, W_bwd, wsW);
  bilstm_scan_kernel<<<16, THREADS, 0, stream>>>(x, wsW, b_fwd, b_bwd, out);
}
